// MultiHeadAttention_1924145348806
// MI455X (gfx1250) — compile-verified, matrix operand order checked by reading, not run on hardware
//
#include <hip/hip_runtime.h>
#include <hip/hip_bf16.h>

// ---------- types ----------
typedef __bf16 bf16;
typedef __attribute__((ext_vector_type(16))) __bf16 v16bf;
typedef __attribute__((ext_vector_type(8)))  __bf16 v8bf;
typedef __attribute__((ext_vector_type(8)))  float   v8f;
typedef int v4i_t __attribute__((vector_size(16)));   // matches builtin proto

static constexpr int Bc = 2, Tc = 2048, Ec = 2048, Hc = 16, Dc = 128;

// ---------- CDNA5 async global->LDS path (probe via __has_builtin) ----------
#if defined(__gfx1250__) && __has_builtin(__builtin_amdgcn_global_load_async_to_lds_b128)
#define ASYNC_LDS 1
#else
#define ASYNC_LDS 0
#endif

// 16-byte global->LDS copy. Async (ASYNCcnt-tracked, no VGPR round trip) when
// available; otherwise a plain load+ds_store through registers.
__device__ __forceinline__ void copy16_g2l(const bf16* g, bf16* l) {
#if ASYNC_LDS
  __builtin_amdgcn_global_load_async_to_lds_b128(
      (__attribute__((address_space(1))) v4i_t*)(g),
      (__attribute__((address_space(3))) v4i_t*)(l), 0, 0);
#else
  *(v8bf*)l = *(const v8bf*)g;
#endif
}

__device__ __forceinline__ void wait_async0() {
#if ASYNC_LDS
#if __has_builtin(__builtin_amdgcn_s_wait_asynccnt)
  __builtin_amdgcn_s_wait_asynccnt(0);
#else
  asm volatile("s_wait_asynccnt 0x0" ::: "memory");
#endif
#endif
}

// ---------- helpers ----------
__device__ __forceinline__ bf16 f2bf(float f) {
  union { float f; unsigned u; } in; in.f = f;
  unsigned u = in.u;
  unsigned r = u + 0x7FFFu + ((u >> 16) & 1u);   // round-to-nearest-even
  union { unsigned short s; bf16 b; } out; out.s = (unsigned short)(r >> 16);
  return out.b;
}

union FragU { v16bf v; v8bf h[2]; };

// A/B fragment for v_wmma_*_16x16x32_bf16: per-lane two contiguous 16B chunks
// at kbase and kbase+16 within the 32-wide K tile (kbase = (lane>>4)*8).
__device__ __forceinline__ v16bf load_frag(const bf16* p, int kbase) {
  FragU f;
  f.h[0] = *(const v8bf*)(p + kbase);
  f.h[1] = *(const v8bf*)(p + kbase + 16);
  return f.v;
}

__device__ __forceinline__ v8f wmma_bf16(v16bf a, v16bf b, v8f c) {
  return __builtin_amdgcn_wmma_f32_16x16x32_bf16(
      /*neg_a=*/false, a, /*neg_b=*/false, b,
      /*c_mod=*/(short)0, c, /*reuse_a=*/false, /*reuse_b=*/false);
}

// ================= f32 -> bf16 pack (one-time, bandwidth-trivial) ==========
__global__ void __launch_bounds__(256)
cvt_bf16_kernel(const float* __restrict__ src, bf16* __restrict__ dst, int n8) {
  const int i = blockIdx.x * 256 + threadIdx.x;
  if (i >= n8) return;
  const float4* s = (const float4*)src + (size_t)i * 2;
  const float4 f0 = s[0], f1 = s[1];
  union { v8bf v; bf16 e[8]; } pk;
  pk.e[0] = f2bf(f0.x); pk.e[1] = f2bf(f0.y); pk.e[2] = f2bf(f0.z); pk.e[3] = f2bf(f0.w);
  pk.e[4] = f2bf(f1.x); pk.e[5] = f2bf(f1.y); pk.e[6] = f2bf(f1.z); pk.e[7] = f2bf(f1.w);
  *((v8bf*)dst + i) = pk.v;
}

// ================= GEMM (NT): C[M,N] = A[M,K] * B[N,K]^T ===================
// A, B bf16 row-major. Double-buffered K-loop: async copies for step k+1 are
// issued right after the barrier and overlap the WMMA work on step k.
// Block tile 128x128, K-step 32, 256 threads = 8 waves, wave tile 32x64.
template <bool BF16OUT>
__global__ void __launch_bounds__(256)
gemm_nt_kernel(const bf16* __restrict__ A, const bf16* __restrict__ Bm,
               bf16* __restrict__ Cb, float* __restrict__ Cf,
               const float* __restrict__ bias, int M, int N, int Kd) {
  __shared__ bf16 ldsA[2][128 * 40];  // 128 rows x 32 (+8 pad) bf16, x2 buf
  __shared__ bf16 ldsB[2][128 * 40];

  const int tid  = threadIdx.x;
  const int lane = tid & 31;
  const int wv   = tid >> 5;
  const int bM   = blockIdx.y * 128;
  const int bN   = blockIdx.x * 128;
  const int wM   = (wv & 3) * 32;
  const int wN   = (wv >> 2) * 64;
  const int lm   = lane & 15;
  const int kb8  = (lane >> 4) * 8;  // k-chunk base; also M-half in C layout

  const v8f vz = {0.f, 0.f, 0.f, 0.f, 0.f, 0.f, 0.f, 0.f};
  v8f acc[2][4];
#pragma unroll
  for (int i = 0; i < 2; ++i)
#pragma unroll
    for (int j = 0; j < 4; ++j) acc[i][j] = vz;

  auto issue_tiles = [&](int k0, int buf) {
#pragma unroll
    for (int c = 0; c < 2; ++c) {
      const int cid = tid + c * 256;     // 512 16B chunks per 128x32 tile
      const int r   = cid >> 2;
      const int cc  = (cid & 3) * 8;
      copy16_g2l(&A [(size_t)(bM + r) * Kd + k0 + cc], &ldsA[buf][r * 40 + cc]);
      copy16_g2l(&Bm[(size_t)(bN + r) * Kd + k0 + cc], &ldsB[buf][r * 40 + cc]);
    }
  };

  issue_tiles(0, 0);
  int cur = 0;
  for (int k0 = 0; k0 < Kd; k0 += 32) {
    wait_async0();
    __syncthreads();                       // tiles in buf `cur` visible to all
    if (k0 + 32 < Kd) issue_tiles(k0 + 32, cur ^ 1);  // overlap with WMMA

    const bf16* baseA = ldsA[cur];
    const bf16* baseB = ldsB[cur];
    const v16bf a0 = load_frag(&baseA[(wM + lm) * 40], kb8);
    const v16bf a1 = load_frag(&baseA[(wM + 16 + lm) * 40], kb8);
#pragma unroll
    for (int j = 0; j < 4; ++j) {
      const v16bf bb = load_frag(&baseB[(wN + j * 16 + lm) * 40], kb8);
      acc[0][j] = wmma_bf16(a0, bb, acc[0][j]);
      acc[1][j] = wmma_bf16(a1, bb, acc[1][j]);
    }
    cur ^= 1;
  }

  // C/D layout: VGPR r -> M = r + 8*(lane>=16); N = lane&15 within 16x16 tile.
#pragma unroll
  for (int i = 0; i < 2; ++i)
#pragma unroll
    for (int j = 0; j < 4; ++j) {
      const int col = bN + wN + j * 16 + lm;
#pragma unroll
      for (int r = 0; r < 8; ++r) {
        const int row = bM + wM + i * 16 + kb8 + r;
        const float v = acc[i][j][r];
        if (BF16OUT) Cb[(size_t)row * N + col] = f2bf(v);
        else         Cf[(size_t)row * N + col] = v + bias[col];
      }
    }
}

// ================= Flash attention (causal), per (b,h) =====================
// Q-block 64 (4 waves x 16 rows), key-block 64, D = 128, all WMMA bf16.
__global__ void __launch_bounds__(128)
attn_kernel(const bf16* __restrict__ Q, const bf16* __restrict__ K,
            const bf16* __restrict__ V, bf16* __restrict__ ctx) {
  __shared__ bf16 ldsQ[64 * 136];   // [q][d]
  __shared__ bf16 ldsK[64 * 136];   // [key][d]
  __shared__ bf16 ldsVt[128 * 72];  // [d][key]  (transposed for B-fragments)
  __shared__ bf16 ldsP[64 * 72];    // [q][key]  softmax probs as bf16

  const int tid  = threadIdx.x;
  const int lane = tid & 31;
  const int wv   = tid >> 5;        // 4 waves, wave handles rows wv*16..+15
  const int lm   = lane & 15;
  const int kb8  = (lane >> 4) * 8;

  const int qb = blockIdx.x;        // query block of 64
  const int b  = blockIdx.y / Hc;
  const int h  = blockIdx.y % Hc;
  // faithful reshape(b,h,t,d) view of the flat (b, t, e) GEMM output:
  const size_t headOff = (size_t)b * Tc * Ec + (size_t)h * Tc * Dc;
  const bf16* Qg = Q + headOff;
  const bf16* Kg = K + headOff;
  const bf16* Vg = V + headOff;

  // issue Q tile copies (64 x 128); consumed at the first in-loop wait+barrier
#pragma unroll
  for (int c = 0; c < 8; ++c) {
    const int cid = tid + c * 128;
    const int r   = cid >> 4;
    const int cc  = (cid & 15) * 8;
    copy16_g2l(&Qg[(size_t)(qb * 64 + r) * Dc + cc], &ldsQ[r * 136 + cc]);
  }

  const v8f vz = {0.f, 0.f, 0.f, 0.f, 0.f, 0.f, 0.f, 0.f};
  v8f accO[8];
#pragma unroll
  for (int j = 0; j < 8; ++j) accO[j] = vz;
  float mrow[8], lrow[8];
#pragma unroll
  for (int r = 0; r < 8; ++r) { mrow[r] = -1e30f; lrow[r] = 0.f; }

  const float scl = 0.08838834764831845f;  // 1/sqrt(128)
  const int nkb = qb + 1;                  // causal: key blocks <= query block

  for (int kb = 0; kb < nkb; ++kb) {
    __syncthreads();                       // prior S/P@V reads of LDS done
    const int t0 = kb * 64;
#pragma unroll
    for (int c = 0; c < 8; ++c) {
      const int cid = tid + c * 128;
      const int r   = cid >> 4;
      const int cc  = (cid & 15) * 8;
      copy16_g2l(&Kg[(size_t)(t0 + r) * Dc + cc], &ldsK[r * 136 + cc]);
      union { v8bf v; bf16 e[8]; } tv;   // V staged transposed -> manual path
      tv.v = *(const v8bf*)&Vg[(size_t)(t0 + r) * Dc + cc];
#pragma unroll
      for (int i = 0; i < 8; ++i) ldsVt[(cc + i) * 72 + r] = tv.e[i];
    }
    if (kb + 1 < nkb) {                    // prefetch next K/V tiles into GL2
      const size_t nt = (size_t)(t0 + 64) * Dc;
      __builtin_prefetch(&Kg[nt] + tid * 64, 0, 1);
      __builtin_prefetch(&Vg[nt] + tid * 64, 0, 1);
    }
    wait_async0();
    __syncthreads();

    // ---- S = Q K^T : 16 x 64 per wave (4 tiles), K-dim = D = 128 ----
    v8f accS[4];
#pragma unroll
    for (int j = 0; j < 4; ++j) accS[j] = vz;
#pragma unroll
    for (int ks = 0; ks < 4; ++ks) {
      const v16bf a = load_frag(&ldsQ[(wv * 16 + lm) * 136 + ks * 32], kb8);
#pragma unroll
      for (int j = 0; j < 4; ++j) {
        const v16bf bb = load_frag(&ldsK[(j * 16 + lm) * 136 + ks * 32], kb8);
        accS[j] = wmma_bf16(a, bb, accS[j]);
      }
    }

    // ---- scale + causal mask ----
#pragma unroll
    for (int j = 0; j < 4; ++j) {
      const int kg = t0 + j * 16 + lm;
#pragma unroll
      for (int r = 0; r < 8; ++r) {
        const int qg = qb * 64 + wv * 16 + kb8 + r;
        const float s = accS[j][r] * scl;
        accS[j][r] = (kg <= qg) ? s : -1e30f;
      }
    }

    // ---- online softmax (row stats within a 16-lane half) ----
#pragma unroll
    for (int r = 0; r < 8; ++r) {
      float mx = -1e30f;
#pragma unroll
      for (int j = 0; j < 4; ++j) mx = fmaxf(mx, accS[j][r]);
      mx = fmaxf(mx, __shfl_xor(mx, 1, 32));
      mx = fmaxf(mx, __shfl_xor(mx, 2, 32));
      mx = fmaxf(mx, __shfl_xor(mx, 4, 32));
      mx = fmaxf(mx, __shfl_xor(mx, 8, 32));
      const float mn    = fmaxf(mrow[r], mx);
      const float alpha = __expf(mrow[r] - mn);
      mrow[r] = mn;
      float rs = 0.f;
#pragma unroll
      for (int j = 0; j < 4; ++j) {
        const float p = __expf(accS[j][r] - mn);
        accS[j][r] = p;
        rs += p;
      }
      rs += __shfl_xor(rs, 1, 32);
      rs += __shfl_xor(rs, 2, 32);
      rs += __shfl_xor(rs, 4, 32);
      rs += __shfl_xor(rs, 8, 32);
      lrow[r] = lrow[r] * alpha + rs;
#pragma unroll
      for (int j = 0; j < 8; ++j) accO[j][r] *= alpha;
    }

    // ---- P (C-layout) -> LDS -> A-layout fragments ----
#pragma unroll
    for (int j = 0; j < 4; ++j)
#pragma unroll
      for (int r = 0; r < 8; ++r)
        ldsP[(wv * 16 + kb8 + r) * 72 + j * 16 + lm] = f2bf(accS[j][r]);

    __syncthreads();

    // ---- O += P V : K-dim = 64 keys (2 steps), 8 d-tiles ----
#pragma unroll
    for (int ks = 0; ks < 2; ++ks) {
      const v16bf a = load_frag(&ldsP[(wv * 16 + lm) * 72 + ks * 32], kb8);
#pragma unroll
      for (int j = 0; j < 8; ++j) {
        const v16bf bb = load_frag(&ldsVt[(j * 16 + lm) * 72 + ks * 32], kb8);
        accO[j] = wmma_bf16(a, bb, accO[j]);
      }
    }
  }

  // ---- normalize + store ctx (bf16) in (b, t, h*D + d) layout ----
#pragma unroll
  for (int j = 0; j < 8; ++j)
#pragma unroll
    for (int r = 0; r < 8; ++r) {
      const int qg = qb * 64 + wv * 16 + kb8 + r;
      const int d  = j * 16 + lm;
      ctx[(size_t)b * Tc * Ec + (size_t)qg * Ec + h * Dc + d] =
          f2bf(accO[j][r] / lrow[r]);
    }
}

// ================= launch ==================================================
extern "C" void kernel_launch(void* const* d_in, const int* in_sizes, int n_in,
                              void* d_out, int out_size, void* d_ws,
                              size_t ws_size, hipStream_t stream) {
  (void)in_sizes; (void)n_in; (void)out_size; (void)ws_size;
  const float* x  = (const float*)d_in[0];
  const float* Wq = (const float*)d_in[1];
  const float* Wk = (const float*)d_in[2];
  const float* Wv = (const float*)d_in[3];
  const float* Wp = (const float*)d_in[4];
  const float* bp = (const float*)d_in[5];

  const size_t XN = (size_t)Bc * Tc * Ec;  // 8,388,608
  const size_t WN = (size_t)Ec * Ec;       // 4,194,304

  bf16* xb   = (bf16*)d_ws;
  bf16* Wqb  = xb  + XN;
  bf16* Wkb  = Wqb + WN;
  bf16* Wvb  = Wkb + WN;
  bf16* Wpb  = Wvb + WN;
  bf16* Qb   = Wpb + WN;
  bf16* Kb   = Qb  + XN;
  bf16* Vb   = Kb  + XN;
  bf16* ctxb = Vb  + XN;

  // one-time f32 -> bf16 packs (bandwidth-trivial vs 23.3 TB/s)
  const int x8 = (int)(XN / 8), w8 = (int)(WN / 8);
  cvt_bf16_kernel<<<(x8 + 255) / 256, 256, 0, stream>>>(x,  xb,  x8);
  cvt_bf16_kernel<<<(w8 + 255) / 256, 256, 0, stream>>>(Wq, Wqb, w8);
  cvt_bf16_kernel<<<(w8 + 255) / 256, 256, 0, stream>>>(Wk, Wkb, w8);
  cvt_bf16_kernel<<<(w8 + 255) / 256, 256, 0, stream>>>(Wv, Wvb, w8);
  cvt_bf16_kernel<<<(w8 + 255) / 256, 256, 0, stream>>>(Wp, Wpb, w8);

  const int M = Bc * Tc;       // 4096
  dim3 gg(Ec / 128, M / 128);  // (16, 32)

  gemm_nt_kernel<true><<<gg, 256, 0, stream>>>(xb, Wqb, Qb, nullptr, nullptr, M, Ec, Ec);
  gemm_nt_kernel<true><<<gg, 256, 0, stream>>>(xb, Wkb, Kb, nullptr, nullptr, M, Ec, Ec);
  gemm_nt_kernel<true><<<gg, 256, 0, stream>>>(xb, Wvb, Vb, nullptr, nullptr, M, Ec, Ec);

  attn_kernel<<<dim3(Tc / 64, Bc * Hc), 128, 0, stream>>>(Qb, Kb, Vb, ctxb);

  gemm_nt_kernel<false><<<gg, 256, 0, stream>>>(ctxb, Wpb, nullptr, (float*)d_out, bp, M, Ec, Ec);
}
